// FP32ToFP8Converter_23407571764187
// MI455X (gfx1250) — matrix operand order repaired and checked by reading, not compiled, not run on hardware
//
#include <hip/hip_runtime.h>

// CDNA5 / gfx1250: wave32, WMMA 16x16x32 bf16 with f32 accumulation.
typedef __attribute__((ext_vector_type(16))) __bf16 v16bf;
typedef __attribute__((ext_vector_type(8)))  float  v8f;

union AB16 {
    unsigned int u[8];
    v16bf v;
};

// {bf16(fodd) : bf16(feven)} — bf16 of 0.0f/1.0f is exactly the top 16 bits.
__device__ __forceinline__ unsigned int pack2bf16(unsigned int feven, unsigned int fodd) {
    return __builtin_amdgcn_perm(fodd, feven, 0x07060302u);
}

// FP32 fields -> FP8 E4M3 fields. Fully branchless: all paths computed,
// results blended with -(cond) masks so no EXEC-divergent control flow is
// generated (exp>134 holds for ~half the lanes on random input, so an
// if-converted branch would serialize both sides every wave).
__device__ __forceinline__ void fp8_fields(unsigned int exp, unsigned int mant,
                                           unsigned int& exp8, unsigned int& mant8) {
    // ---- normal path: RNE round 23 -> 3 mantissa bits ----
    const unsigned int kept  = mant >> 20;
    const unsigned int R     = (mant >> 19) & 1u;
    const unsigned int S     = (unsigned int)((mant & 0x7FFFFu) != 0u);
    const unsigned int mr    = kept + (R & (S | (kept & 1u)));
    const unsigned int carry = mr >> 3;
    // carry==1 -> mantissa cleared; blend via mask instead of select
    const unsigned int m_carry   = (unsigned int)-(int)carry;     // 0 or ~0
    const unsigned int mant_norm = (mr & 7u) & ~m_carry;
    const unsigned int exp_norm  = exp - 120u + carry;            // only used when exp>=121

    // ---- subnormal path: shift 1.mant right by (141-exp), RNE ----
    const unsigned int full = (1u << 23) | mant;
    int shi = 141 - (int)exp;
    shi = shi < 1 ? 1 : (shi > 24 ? 24 : shi);
    const unsigned int sh     = (unsigned int)shi;
    const unsigned int kept_s = full >> sh;
    const unsigned int Rs     = (full >> (sh - 1u)) & 1u;
    const unsigned int Ss     = (unsigned int)((full & ((1u << (sh - 1u)) - 1u)) != 0u);
    const unsigned int ms     = kept_s + (Rs & (Ss | (kept_s & 1u)));
    const unsigned int m_subof   = (unsigned int)-(int)(ms >= 8u);
    const unsigned int sub_exp   = 1u & m_subof;
    const unsigned int sub_mant  = ms & ~m_subof;

    // ---- branchless 4-way select: ovf > sub > und > normal ----
    const unsigned int m_ovf  = (unsigned int)-(int)(exp > 134u);
    const unsigned int m_sub  = (unsigned int)-(int)((exp >= 117u) & (exp <= 120u));
    const unsigned int m_und  = (unsigned int)-(int)(exp < 117u);
    const unsigned int m_norm = ~(m_ovf | m_sub | m_und);
    exp8  = (m_ovf & 15u) | (m_sub & sub_exp)  | (m_norm & exp_norm);
    mant8 = (m_ovf & 6u)  | (m_sub & sub_mant) | (m_norm & mant_norm);
}

// One wave handles 32 values per chunk via two chained bf16 WMMAs.
//  A1 rows 0/1/2  = sign/exp/mant weights (group 1, D lanes 0-15)
//  A2 rows 8/9/10 = sign/exp/mant weights (group 2, D lanes 16-31)
//  B  columns     = 32 bit-pulses of one value each.
// All products are 0 or powers of two <= 2^22 (exact in bf16); f32
// accumulation of sums < 2^24 is exact, so cvt-to-int is lossless.
__global__ void __launch_bounds__(256)
fp8_pulse_wmma_kernel(const float* __restrict__ in, float* __restrict__ out, int nvals)
{
    const int lane = threadIdx.x & 31;
    const int row  = lane & 15;     // A-matrix row / B-matrix column owned by lane
    const int hi   = lane >> 4;     // 0: K=0..15 half, 1: K=16..31 half
    const int wavesPerBlock = blockDim.x >> 5;
    const int gwave   = blockIdx.x * wavesPerBlock + (threadIdx.x >> 5);
    const int nwaves  = gridDim.x * wavesPerBlock;
    const int nchunks = nvals >> 5;                 // 32 values per wave-chunk

    // ---- build the two constant A operands (bit weights, exact powers of 2) ----
    AB16 a1, a2;
#pragma unroll
    for (int j = 0; j < 8; ++j) { a1.u[j] = 0u; a2.u[j] = 0u; }
#pragma unroll
    for (int j = 0; j < 16; ++j) {
        // A in contiguous k order (code's choice): lanes 0-15 hold K=0..15, lanes 16-31 K=16..31 (matches B)
        const int k = (hi << 4) + j;
        const unsigned int wsign = (k == 0) ? 0x3F80u : 0u;                                   // 1.0
        const unsigned int wexp  = (k >= 1 && k <= 8) ? (unsigned int)((127 + (8  - k)) << 7) : 0u; // 2^(8-k)
        const unsigned int wmant = (k >= 9)           ? (unsigned int)((127 + (31 - k)) << 7) : 0u; // 2^(31-k)
        const unsigned int w1 = (row == 0) ? wsign : (row == 1) ? wexp : (row == 2)  ? wmant : 0u;
        const unsigned int w2 = (row == 8) ? wsign : (row == 9) ? wexp : (row == 10) ? wmant : 0u;
        a1.u[j >> 1] |= w1 << ((j & 1) * 16);
        a2.u[j >> 1] |= w2 << ((j & 1) * 16);
    }

    for (int c = gwave; c < nchunks; c += nwaves) {
        const int vbase = c << 5;
        // B, same contiguous k order as A: lanes 0-15 hold K=0..15 of column `row`,
        // lanes 16-31 hold K=16..31 -> each lane loads 16 consecutive floats.
        const float* p1 = in + (((size_t)(vbase + row)) << 5) + (hi << 4); // group 1: values vbase+0..15
        const float* p2 = p1 + (16u << 5);                                // group 2: values vbase+16..31

        const uint4* q1 = (const uint4*)p1;
        const uint4* q2 = (const uint4*)p2;
        uint4 x0 = q1[0], x1 = q1[1], x2 = q1[2], x3 = q1[3];
        uint4 y0 = q2[0], y1 = q2[1], y2 = q2[2], y3 = q2[3];

        // speculative prefetch of the next chunk for this wave (global_prefetch_b8)
        const size_t stride = ((size_t)nwaves) << 10;   // nwaves * 32 values * 32 floats
        __builtin_prefetch(p1 + stride, 0, 1);
        __builtin_prefetch(p2 + stride, 0, 1);

        AB16 b1, b2;
        b1.u[0] = pack2bf16(x0.x, x0.y); b1.u[1] = pack2bf16(x0.z, x0.w);
        b1.u[2] = pack2bf16(x1.x, x1.y); b1.u[3] = pack2bf16(x1.z, x1.w);
        b1.u[4] = pack2bf16(x2.x, x2.y); b1.u[5] = pack2bf16(x2.z, x2.w);
        b1.u[6] = pack2bf16(x3.x, x3.y); b1.u[7] = pack2bf16(x3.z, x3.w);
        b2.u[0] = pack2bf16(y0.x, y0.y); b2.u[1] = pack2bf16(y0.z, y0.w);
        b2.u[2] = pack2bf16(y1.x, y1.y); b2.u[3] = pack2bf16(y1.z, y1.w);
        b2.u[4] = pack2bf16(y2.x, y2.y); b2.u[5] = pack2bf16(y2.z, y2.w);
        b2.u[6] = pack2bf16(y3.x, y3.y); b2.u[7] = pack2bf16(y3.z, y3.w);

        // D = A1*B1; D = A2*B2 + D   (disjoint rows -> lanes 0-31 all carry results)
        v8f acc = {};
        acc = __builtin_amdgcn_wmma_f32_16x16x32_bf16(false, a1.v, false, b1.v,
                                                      (short)0, acc, false, false);
        acc = __builtin_amdgcn_wmma_f32_16x16x32_bf16(false, a2.v, false, b2.v,
                                                      (short)0, acc, false, false);

        // lane L owns value vbase+L: D VGPR0=sign, VGPR1=exp, VGPR2=mant (exact f32 ints)
        const float        s_f  = acc[0];
        const unsigned int exp  = (unsigned int)acc[1];
        const unsigned int mant = (unsigned int)acc[2];
        unsigned int exp8, mant8;
        fp8_fields(exp, mant, exp8, mant8);

        float4 o0, o1;
        o0.x = s_f;                              // sign pulse is already 0.0/1.0
        o0.y = (float)((exp8 >> 3) & 1u);
        o0.z = (float)((exp8 >> 2) & 1u);
        o0.w = (float)((exp8 >> 1) & 1u);
        o1.x = (float)( exp8        & 1u);
        o1.y = (float)((mant8 >> 2) & 1u);
        o1.z = (float)((mant8 >> 1) & 1u);
        o1.w = (float)( mant8       & 1u);

        float4* op = (float4*)(out + (((size_t)(vbase + lane)) << 3));
        op[0] = o0;
        op[1] = o1;
    }
}

// Scalar tail for nvals not divisible by 32 (not hit for N=2^21, kept for safety).
__global__ void fp8_pulse_tail_kernel(const float* __restrict__ in, float* __restrict__ out,
                                      int start, int nvals)
{
    const int v = start + blockIdx.x * blockDim.x + threadIdx.x;
    if (v >= nvals) return;
    const unsigned int* p = (const unsigned int*)(in + ((size_t)v << 5));
    // pulses are exactly 0.0f/1.0f -> bit 23 of the f32 pattern is the bit value
    const unsigned int s = (p[0] >> 23) & 1u;
    unsigned int exp = 0u, mant = 0u;
#pragma unroll
    for (int i = 1; i < 9; ++i)  exp  = (exp  << 1) | ((p[i] >> 23) & 1u);
#pragma unroll
    for (int i = 9; i < 32; ++i) mant = (mant << 1) | ((p[i] >> 23) & 1u);
    unsigned int exp8, mant8;
    fp8_fields(exp, mant, exp8, mant8);
    float* o = out + ((size_t)v << 3);
    o[0] = (float)s;
    o[1] = (float)((exp8 >> 3) & 1u);
    o[2] = (float)((exp8 >> 2) & 1u);
    o[3] = (float)((exp8 >> 1) & 1u);
    o[4] = (float)( exp8        & 1u);
    o[5] = (float)((mant8 >> 2) & 1u);
    o[6] = (float)((mant8 >> 1) & 1u);
    o[7] = (float)( mant8       & 1u);
}

extern "C" void kernel_launch(void* const* d_in, const int* in_sizes, int n_in,
                              void* d_out, int out_size, void* d_ws, size_t ws_size,
                              hipStream_t stream) {
    const float* in  = (const float*)d_in[0];
    float*       out = (float*)d_out;
    const int nfloats = in_sizes[0];
    const int nvals   = nfloats / 32;     // 2,097,152 for the reference setup
    const int nchunks = nvals / 32;       // 32 values per wave

    if (nchunks > 0) {
        const int threads = 256;          // 8 waves per block on wave32
        const int wpb = threads / 32;
        int blocks = (nchunks + wpb - 1) / wpb;
        if (blocks > 2048) blocks = 2048; // grid-stride: ~4 chunks/wave amortizes A setup
        fp8_pulse_wmma_kernel<<<blocks, threads, 0, stream>>>(in, out, nvals);
    }
    const int rem = nvals - nchunks * 32;
    if (rem > 0) {
        fp8_pulse_tail_kernel<<<(rem + 255) / 256, 256, 0, stream>>>(in, out, nchunks * 32, nvals);
    }
}
